// InfoNCELoss_89919435309435
// MI455X (gfx1250) — matrix operand order repaired and checked by reading, not compiled, not run on hardware
//
#include <hip/hip_runtime.h>
#include <math.h>

// SupCon / InfoNCE loss, fused: per-row (Z, S, C) statistics computed tile-by-tile
// with v_wmma_f32_16x16x32_f16, never materializing the 8192x8192 sim matrix.
//
//   t_i  = (S_i - C_i * log(Z_i + eps)) / (C_i + eps)
//   loss = -mean_i t_i

#define TEMP_INV 14.2857142857142857f  // 1 / 0.07
#define EPS_F    1e-8f

typedef __attribute__((ext_vector_type(16))) _Float16 v16h;
typedef __attribute__((ext_vector_type(8)))  _Float16 v8h;
typedef __attribute__((ext_vector_type(4)))  _Float16 v4h;
typedef __attribute__((ext_vector_type(8)))  float    v8f;

union V16 { v16h v; v8h h[2]; };

// ---------------------------------------------------------------------------
// Kernel 1: L2-normalize rows (f32 [N][128]) -> fp16 [N][128] in workspace.
// One wave32 per row: lane loads float4 (4*32 = 128 elements).
// ---------------------------------------------------------------------------
__global__ __launch_bounds__(256)
void normalize_rows_f16(const float* __restrict__ feat,
                        _Float16* __restrict__ fh, int N) {
  const int wavesPerBlock = blockDim.x >> 5;
  const int row  = blockIdx.x * wavesPerBlock + (threadIdx.x >> 5);
  const int lane = threadIdx.x & 31;
  if (row >= N) return;

  const float4* p = (const float4*)(feat + (size_t)row * 128);
  float4 x = p[lane];
  float ss = x.x * x.x + x.y * x.y + x.z * x.z + x.w * x.w;
#pragma unroll
  for (int m = 16; m >= 1; m >>= 1) ss += __shfl_xor(ss, m, 32);

  const float scale = 1.0f / fmaxf(sqrtf(ss), 1e-12f);
  v4h o;
  o.x = (_Float16)(x.x * scale);
  o.y = (_Float16)(x.y * scale);
  o.z = (_Float16)(x.z * scale);
  o.w = (_Float16)(x.w * scale);
  ((v4h*)(fh + (size_t)row * 128))[lane] = o;
}

// ---------------------------------------------------------------------------
// Kernel 2: fused sim-GEMM + row statistics.
// One wave per 16-row strip. For each 16-col tile: 4x v_wmma_f32_16x16x32_f16
// (K = 128), then epilogue accumulates Z/S/C per row in registers.
//
// WMMA operand layouts (CDNA5 ISA 7.12.2, wave32):
//  A (16x32 f16):  lane L holds row M = L&15.
//    halves 0..7  -> K = kk*32 + (L>=16 ? 8 : 0) + h        (VGPR0-3)
//    halves 8..15 -> K = kk*32 + 16 + (L>=16 ? 8 : 0) + h-8 (VGPR4-7)
//  B (32x16 f16):  lane L holds col N = L&15.
//    halves 0..15 -> same K mapping as A above
//  C/D (16x16 f32): VGPR v: lanes 0-15 -> M=v, lanes 16-31 -> M=v+8; N=L&15.
// ---------------------------------------------------------------------------
__global__ __launch_bounds__(256)
void supcon_rowstats(const _Float16* __restrict__ fh,
                     const int* __restrict__ labels,
                     float* __restrict__ rowterm, int N) {
  const int lane  = threadIdx.x & 31;
  const int wave  = threadIdx.x >> 5;
  const int strip = blockIdx.x * (blockDim.x >> 5) + wave;
  const int ibase = strip << 4;
  const int hi    = lane >> 4;     // 0 for lanes 0-15, 1 for lanes 16-31
  const int l15   = lane & 15;

  // ---- A tile: rows ibase..ibase+15, all K=128, resident for whole strip ----
  v16h a[4];
  {
    const _Float16* arow = fh + (size_t)(ibase + l15) * 128;
#pragma unroll
    for (int kk = 0; kk < 4; ++kk) {
      const int k0 = kk * 32 + hi * 8;
      V16 t;
      t.h[0] = *(const v8h*)(arow + k0);        // K = k0   .. k0+7
      t.h[1] = *(const v8h*)(arow + k0 + 16);   // K = k0+16.. k0+23
      a[kk] = t.v;
    }
  }

  int   rowLab[8];
  float zAcc[8], sAcc[8], cAcc[8];
#pragma unroll
  for (int v = 0; v < 8; ++v) {
    rowLab[v] = labels[ibase + v + hi * 8];
    zAcc[v] = 0.0f; sAcc[v] = 0.0f; cAcc[v] = 0.0f;
  }

  const int njt = N >> 4;
  for (int jt = 0; jt < njt; ++jt) {
    const int col = (jt << 4) + l15;
    const _Float16* brow = fh + (size_t)col * 128;
    const int colLab = labels[col];

    v8f c = {};
#pragma unroll
    for (int kk = 0; kk < 4; ++kk) {
      const int k0 = kk * 32 + hi * 8;
      V16 t;
      t.h[0] = *(const v8h*)(brow + k0);        // K = k0   .. k0+7
      t.h[1] = *(const v8h*)(brow + k0 + 16);   // K = k0+16.. k0+23
      c = __builtin_amdgcn_wmma_f32_16x16x32_f16(
              /*neg_a=*/false, a[kk], /*neg_b=*/false, t.v,
              /*c_mod=*/(short)0, c, /*reuse_a=*/false, /*reuse_b=*/false);
    }

#pragma unroll
    for (int v = 0; v < 8; ++v) {
      const int   row  = ibase + v + hi * 8;
      const float sim  = c[v] * TEMP_INV;
      const bool  self = (row == col);
      zAcc[v] += self ? 0.0f : __expf(sim);
      const bool pos = (rowLab[v] == colLab) && !self;
      sAcc[v] += pos ? sim : 0.0f;
      cAcc[v] += pos ? 1.0f : 0.0f;
    }
  }

  // Butterfly reduce across the 16 lanes of each half-wave (masks < 16 never
  // cross the two halves, which hold disjoint row sets).
#pragma unroll
  for (int v = 0; v < 8; ++v) {
#pragma unroll
    for (int m = 8; m >= 1; m >>= 1) {
      zAcc[v] += __shfl_xor(zAcc[v], m, 32);
      sAcc[v] += __shfl_xor(sAcc[v], m, 32);
      cAcc[v] += __shfl_xor(cAcc[v], m, 32);
    }
  }

  if (l15 == 0) {
#pragma unroll
    for (int v = 0; v < 8; ++v) {
      const int row = ibase + v + hi * 8;
      rowterm[row] =
          (sAcc[v] - cAcc[v] * __logf(zAcc[v] + EPS_F)) / (cAcc[v] + EPS_F);
    }
  }
}

// ---------------------------------------------------------------------------
// Kernel 3: deterministic single-block reduction -> loss = -sum(t)/N.
// ---------------------------------------------------------------------------
__global__ __launch_bounds__(256)
void reduce_loss(const float* __restrict__ rowterm, float* __restrict__ out,
                 int N) {
  __shared__ float sh[256];
  float s = 0.0f;
  for (int i = threadIdx.x; i < N; i += 256) s += rowterm[i];
  sh[threadIdx.x] = s;
  __syncthreads();
#pragma unroll
  for (int w = 128; w >= 1; w >>= 1) {
    if ((int)threadIdx.x < w) sh[threadIdx.x] += sh[threadIdx.x + w];
    __syncthreads();
  }
  if (threadIdx.x == 0) out[0] = -sh[0] / (float)N;
}

// ---------------------------------------------------------------------------
extern "C" void kernel_launch(void* const* d_in, const int* in_sizes, int n_in,
                              void* d_out, int out_size, void* d_ws,
                              size_t ws_size, hipStream_t stream) {
  const float* feat   = (const float*)d_in[0];
  const int*   labels = (const int*)d_in[1];
  const int N = in_sizes[1];          // 8192 labels; D = in_sizes[0]/N = 128

  _Float16* fh      = (_Float16*)d_ws;                       // N*128 fp16 (2 MB)
  float*    rowterm = (float*)((char*)d_ws +
                               (size_t)N * 128 * sizeof(_Float16));

  // 1) normalize: 8 rows (waves) per 256-thread block
  normalize_rows_f16<<<(N + 7) / 8, 256, 0, stream>>>(feat, fh, N);

  // 2) fused WMMA GEMM + row stats: one wave per 16-row strip, 8 waves/block
  const int nstrips = N >> 4;                  // 512
  supcon_rowstats<<<nstrips / 8, 256, 0, stream>>>(fh, labels, rowterm, N);

  // 3) deterministic final reduction
  reduce_loss<<<1, 256, 0, stream>>>(rowterm, (float*)d_out, N);
}
